// MultiHeadAttention_34608846471924
// MI455X (gfx1250) — compile-verified, matrix operand order checked by reading, not run on hardware
//
#include <hip/hip_runtime.h>

// ---------------------------------------------------------------------------
// MI455X (gfx1250, wave32) multi-head attention, all matmuls on
// v_wmma_f32_16x16x32_f16 (f16 A/B, f32 accum). Workload: ~344 GFLOP vs
// ~256MB HBM traffic -> strongly compute-bound, so f16 WMMA + LDS-resident
// tiles, double-buffered staging (1 barrier/iter) + global_prefetch_b8.
// ---------------------------------------------------------------------------

typedef __attribute__((ext_vector_type(16))) _Float16 v16h;
typedef __attribute__((ext_vector_type(8)))  _Float16 v8h;
typedef __attribute__((ext_vector_type(4)))  _Float16 v4h;
typedef __attribute__((ext_vector_type(8)))  float    v8f;

__device__ __forceinline__ v8f vzero8() {
  v8f z;
#pragma unroll
  for (int i = 0; i < 8; ++i) z[i] = 0.0f;
  return z;
}

__device__ __forceinline__ v16h frag_cat(v8h a, v8h b) {
  return __builtin_shufflevector(a, b, 0,1,2,3,4,5,6,7,8,9,10,11,12,13,14,15);
}

__device__ __forceinline__ v8f wmma_f16(v16h a, v16h b, v8f c) {
  // D = A(16x32 f16) x B(32x16 f16) + C(16x16 f32)
  return __builtin_amdgcn_wmma_f32_16x16x32_f16(
      /*neg_a=*/false, a, /*neg_b=*/false, b,
      /*c_mod=*/(short)0, c, /*reuse_a=*/false, /*reuse_b=*/false);
}

// ---------------------------------------------------------------------------
// Elementwise f32 -> f16 conversion (vectorized x4).
// ---------------------------------------------------------------------------
__global__ __launch_bounds__(256) void cvt_f16_kernel(const float* __restrict__ in,
                                                      _Float16* __restrict__ out,
                                                      int n4) {
  int i = blockIdx.x * 256 + threadIdx.x;
  if (i < n4) {
    float4 v = ((const float4*)in)[i];
    v4h h;
    h[0] = (_Float16)v.x; h[1] = (_Float16)v.y;
    h[2] = (_Float16)v.z; h[3] = (_Float16)v.w;
    ((v4h*)out)[i] = h;
  }
}

// ---------------------------------------------------------------------------
// Weight convert + transpose: Wt[n*K + k] = (f16) W[k*N + n]  (32x32 LDS tile)
// ---------------------------------------------------------------------------
__global__ __launch_bounds__(256) void cvt_wT_kernel(const float* __restrict__ W,
                                                     _Float16* __restrict__ Wt,
                                                     int K, int N) {
  __shared__ _Float16 tile[32][33];
  const int n0 = blockIdx.x * 32, k0 = blockIdx.y * 32;
  const int tx = threadIdx.x & 31, ty = threadIdx.x >> 5;  // ty in 0..7
#pragma unroll
  for (int j = 0; j < 32; j += 8)
    tile[ty + j][tx] = (_Float16)W[(size_t)(k0 + ty + j) * N + (n0 + tx)];
  __syncthreads();
#pragma unroll
  for (int j = 0; j < 32; j += 8)
    Wt[(size_t)(n0 + ty + j) * K + (k0 + tx)] = tile[tx][ty + j];
}

// ---------------------------------------------------------------------------
// WMMA GEMM: C[M,N] = scale * A[M,K] * B[K,N], with B given pre-transposed as
// Bt[N,K] so fragments load as contiguous ds_load_b128 pairs.
// 256 threads = 8 waves; block tile 128x128; wave tile 64x32 (8 accumulators).
// Double-buffered LDS (ping-pong): stage tile i+1 while WMMA on tile i,
// single barrier per k-step; prefetch tile i+2 into L2.
// ---------------------------------------------------------------------------
template <typename OUT>
__global__ __launch_bounds__(256, 2) void wmma_gemm_kernel(
    const _Float16* __restrict__ A, const _Float16* __restrict__ Bt,
    OUT* __restrict__ C, int M, int N, int K, float scale) {
  __shared__ __align__(16) _Float16 As[2][128][40];  // 2 x 128 x 32 halves (+8 pad)
  __shared__ __align__(16) _Float16 Bs[2][128][40];

  const int t = threadIdx.x, lane = t & 31, wave = t >> 5;
  const int l15 = lane & 15, g = lane >> 4;
  const int wm = wave >> 2;  // 0..1 -> 64 rows
  const int wn = wave & 3;   // 0..3 -> 32 cols
  const int m0 = blockIdx.y * 128, n0 = blockIdx.x * 128;

  auto stage = [&](int buf, int k0) {
#pragma unroll
    for (int c = t; c < 512; c += 256) {
      const int row = c >> 2, col = (c & 3) * 8;
      *(v8h*)&As[buf][row][col] = *(const v8h*)&A[(size_t)(m0 + row) * K + k0 + col];
      *(v8h*)&Bs[buf][row][col] = *(const v8h*)&Bt[(size_t)(n0 + row) * K + k0 + col];
    }
  };

  v8f acc[4][2];
#pragma unroll
  for (int mi = 0; mi < 4; ++mi)
#pragma unroll
    for (int ni = 0; ni < 2; ++ni) acc[mi][ni] = vzero8();

  const int nks = K / 32;
  stage(0, 0);
  __syncthreads();

  for (int ks = 0; ks < nks; ++ks) {
    const int cur = ks & 1;
    if (ks + 1 < nks) stage(cur ^ 1, (ks + 1) * 32);
    if (ks + 2 < nks) {  // L2 prefetch two steps ahead (global_prefetch_b8)
      const int k2 = (ks + 2) * 32;
      const int prow = t >> 1, pcol = (t & 1) * 16;
      __builtin_prefetch(&A[(size_t)(m0 + prow) * K + k2 + pcol], 0, 1);
      __builtin_prefetch(&Bt[(size_t)(n0 + prow) * K + k2 + pcol], 0, 1);
    }

    v16h bf[2];
#pragma unroll
    for (int ni = 0; ni < 2; ++ni) {
      const _Float16* bp = &Bs[cur][wn * 32 + ni * 16 + l15][0];
      bf[ni] = frag_cat(*(const v8h*)(bp + g * 8), *(const v8h*)(bp + 16 + g * 8));
    }
#pragma unroll
    for (int mi = 0; mi < 4; ++mi) {
      const _Float16* ap = &As[cur][wm * 64 + mi * 16 + l15][0];
      v16h af = frag_cat(*(const v8h*)(ap + g * 8), *(const v8h*)(ap + 16 + g * 8));
#pragma unroll
      for (int ni = 0; ni < 2; ++ni) acc[mi][ni] = wmma_f16(af, bf[ni], acc[mi][ni]);
    }
    __syncthreads();
  }

  // Epilogue: C/D layout -> row = r + 8g, col = lane&15 within 16x16 tile.
#pragma unroll
  for (int mi = 0; mi < 4; ++mi)
#pragma unroll
    for (int ni = 0; ni < 2; ++ni)
#pragma unroll
      for (int r = 0; r < 8; ++r) {
        const int row = m0 + wm * 64 + mi * 16 + r + 8 * g;
        const int col = n0 + wn * 32 + ni * 16 + l15;
        const float v = acc[mi][ni][r] * scale;
        if constexpr (sizeof(OUT) == 2)
          C[(size_t)row * N + col] = (OUT)(_Float16)v;
        else
          C[(size_t)row * N + col] = v;
      }
}

// ---------------------------------------------------------------------------
// Flash attention per (b, h, 128-row q block). 256 threads = 8 waves, each
// wave owns 16 q rows. Scores are computed TRANSPOSED (A = K tile, B = Q^T
// fragments) so exp(S - m) -> P A-fragment conversion is register-local.
// Online softmax across 64-key blocks; context via A=P, B=V (V^T staged).
// K/V^T staging is double-buffered: one barrier per kv block.
// ---------------------------------------------------------------------------
__global__ __launch_bounds__(256, 1) void attn_kernel(
    const _Float16* __restrict__ Q, const _Float16* __restrict__ Km,
    const _Float16* __restrict__ V, _Float16* __restrict__ Ctx) {
  constexpr int S = 1024, Dm = 2048, HD = 128, KB = 64;
  constexpr int NKB = S / KB;
  const int qb = blockIdx.x, h = blockIdx.y, b = blockIdx.z;
  const int t = threadIdx.x, lane = t & 31, wave = t >> 5;
  const int l15 = lane & 15, g = lane >> 4;
  const size_t base = ((size_t)b * S) * Dm + (size_t)h * HD;
  const int q0 = qb * 128;

  __shared__ __align__(16) _Float16 Qs[128][HD + 8];      // 34816 B
  __shared__ __align__(16) _Float16 Ks[2][KB][HD + 8];    // 2 x 17408 B
  __shared__ __align__(16) _Float16 Vts[2][HD][KB + 8];   // 2 x 18432 B (V^T)

  auto stage_kv = [&](int buf, int k0) {
    for (int c = t; c < KB * (HD / 8); c += 256) {   // 1024 chunks
      const int row = c >> 4, col = (c & 15) * 8;
      *(v8h*)&Ks[buf][row][col] = *(const v8h*)&Km[base + (size_t)(k0 + row) * Dm + col];
      v8h v = *(const v8h*)&V[base + (size_t)(k0 + row) * Dm + col];
#pragma unroll
      for (int j = 0; j < 8; ++j) Vts[buf][col + j][row] = v[j];
    }
  };

  // Stage whole 128 x 128 Q block once (resident for the kernel) + first K/V.
  for (int c = t; c < 128 * (HD / 8); c += 256) {
    const int row = c >> 4, col = (c & 15) * 8;
    *(v8h*)&Qs[row][col] = *(const v8h*)&Q[base + (size_t)(q0 + row) * Dm + col];
  }
  stage_kv(0, 0);
  __syncthreads();

  // B-fragments of Q^T (col = q = wave*16 + l15, halves over hd), one per
  // 32-wide hd step; Qs is never overwritten so these live in registers.
  v16h qfrag[4];
  {
    const _Float16* qp = &Qs[wave * 16 + l15][0];
#pragma unroll
    for (int kk = 0; kk < 4; ++kk)
      qfrag[kk] = frag_cat(*(const v8h*)(qp + kk * 32 + g * 8),
                           *(const v8h*)(qp + kk * 32 + 16 + g * 8));
  }

  float m_i = -3.0e38f, l_i = 0.0f;   // valid for q col = lane&15 (both halves)
  v8f o[8];
#pragma unroll
  for (int n8 = 0; n8 < 8; ++n8) o[n8] = vzero8();

  for (int kb = 0; kb < NKB; ++kb) {
    const int cur = kb & 1;
    if (kb + 1 < NKB) stage_kv(cur ^ 1, (kb + 1) * KB);

    // Transposed scores: st[t8](key 16-tile t8, q 16) = K * Q^T over HD=128.
    v8f st[4];
#pragma unroll
    for (int t8 = 0; t8 < 4; ++t8) {
      st[t8] = vzero8();
      const _Float16* kp = &Ks[cur][t8 * 16 + l15][0];
#pragma unroll
      for (int kk = 0; kk < 4; ++kk) {
        v16h af = frag_cat(*(const v8h*)(kp + kk * 32 + g * 8),
                           *(const v8h*)(kp + kk * 32 + 16 + g * 8));
        st[t8] = wmma_f16(af, qfrag[kk], st[t8]);
      }
    }

    // Online softmax over this 64-key block (reduce across key rows:
    // 8 accum regs + lane-half swap via shfl_xor 16).
    float bmax = -3.0e38f;
#pragma unroll
    for (int t8 = 0; t8 < 4; ++t8)
#pragma unroll
      for (int r = 0; r < 8; ++r) bmax = fmaxf(bmax, st[t8][r]);
    bmax = fmaxf(bmax, __shfl_xor(bmax, 16, 32));
    const float mnew = fmaxf(m_i, bmax);
    const float corr = __expf(m_i - mnew);
    float bsum = 0.0f;
#pragma unroll
    for (int t8 = 0; t8 < 4; ++t8)
#pragma unroll
      for (int r = 0; r < 8; ++r) {
        const float p = __expf(st[t8][r] - mnew);
        st[t8][r] = p;
        bsum += p;
      }
    bsum += __shfl_xor(bsum, 16, 32);
    l_i = l_i * corr + bsum;
    m_i = mnew;

    // Rescale running context: row r of accum tile is q = r + 8g; corr for
    // that q lives at lane (r + 8g).
#pragma unroll
    for (int r = 0; r < 8; ++r) {
      const float cr = __shfl(corr, r + 8 * g, 32);
#pragma unroll
      for (int n8 = 0; n8 < 8; ++n8) o[n8][r] *= cr;
    }

    // P A-fragments (16 q x 32 key) — pure register re-pack of st pairs.
    v16h pf[2];
#pragma unroll
    for (int kA = 0; kA < 2; ++kA)
#pragma unroll
      for (int i = 0; i < 8; ++i) {
        pf[kA][i]     = (_Float16)st[2 * kA][i];
        pf[kA][i + 8] = (_Float16)st[2 * kA + 1][i];
      }

    // Context: o(q, hd-tile n8) += P * V.
#pragma unroll
    for (int n8 = 0; n8 < 8; ++n8) {
      const _Float16* vp = &Vts[cur][n8 * 16 + l15][0];
#pragma unroll
      for (int kA = 0; kA < 2; ++kA) {
        v16h bf = frag_cat(*(const v8h*)(vp + kA * 32 + g * 8),
                           *(const v8h*)(vp + kA * 32 + 16 + g * 8));
        o[n8] = wmma_f16(pf[kA], bf, o[n8]);
      }
    }
    __syncthreads();
  }

  // Epilogue: divide by row sum, store f16 context [B,S,H*HD].
#pragma unroll
  for (int r = 0; r < 8; ++r) {
    const float li = __shfl(1.0f / l_i, r + 8 * g, 32);
    const int q = q0 + wave * 16 + r + 8 * g;
    const size_t rb = base + (size_t)q * Dm;
#pragma unroll
    for (int n8 = 0; n8 < 8; ++n8)
      Ctx[rb + n8 * 16 + l15] = (_Float16)(o[n8][r] * li);
  }
}

// ---------------------------------------------------------------------------
// Host-side orchestration (graph-capture safe: launches only).
// ---------------------------------------------------------------------------
extern "C" void kernel_launch(void* const* d_in, const int* in_sizes, int n_in,
                              void* d_out, int out_size, void* d_ws, size_t ws_size,
                              hipStream_t stream) {
  constexpr int B = 8, S = 1024, D = 2048, H = 16, HD = 128;
  constexpr int M = B * S;               // 8192 rows
  const float qscale = 0.08838834764831845f;  // 1/sqrt(HD)

  const float* inq  = (const float*)d_in[0];
  const float* inkv = (const float*)d_in[1];
  const float* Wq   = (const float*)d_in[2];
  const float* Wk   = (const float*)d_in[3];
  const float* Wv   = (const float*)d_in[4];
  const float* Wo   = (const float*)d_in[5];
  float* out = (float*)d_out;

  const size_t SZ_X = (size_t)M * D * sizeof(_Float16);   // 32 MB
  const size_t SZ_W = (size_t)D * D * sizeof(_Float16);   //  8 MB
  if (ws_size < 6 * SZ_X + 4 * SZ_W) return;              // need ~224 MB scratch

  char* p = (char*)d_ws;
  _Float16* Xq  = (_Float16*)(p);
  _Float16* Xkv = (_Float16*)(p + SZ_X);
  _Float16* WqT = (_Float16*)(p + 2 * SZ_X);
  _Float16* WkT = (_Float16*)(p + 2 * SZ_X + 1 * SZ_W);
  _Float16* WvT = (_Float16*)(p + 2 * SZ_X + 2 * SZ_W);
  _Float16* WoT = (_Float16*)(p + 2 * SZ_X + 3 * SZ_W);
  _Float16* Qh  = (_Float16*)(p + 2 * SZ_X + 4 * SZ_W);
  _Float16* Kh  = (_Float16*)(p + 3 * SZ_X + 4 * SZ_W);
  _Float16* Vh  = (_Float16*)(p + 4 * SZ_X + 4 * SZ_W);
  _Float16* Ctx = (_Float16*)(p + 5 * SZ_X + 4 * SZ_W);

  // 1) Convert activations to f16.
  const int n4 = (M * D) / 4;  // 4.19M float4 chunks
  cvt_f16_kernel<<<n4 / 256, 256, 0, stream>>>(inq,  Xq,  n4);
  cvt_f16_kernel<<<n4 / 256, 256, 0, stream>>>(inkv, Xkv, n4);

  // 2) Convert + transpose weights (Wt[N,K]).
  dim3 wgrid(D / 32, D / 32);
  cvt_wT_kernel<<<wgrid, 256, 0, stream>>>(Wq, WqT, D, D);
  cvt_wT_kernel<<<wgrid, 256, 0, stream>>>(Wk, WkT, D, D);
  cvt_wT_kernel<<<wgrid, 256, 0, stream>>>(Wv, WvT, D, D);
  cvt_wT_kernel<<<wgrid, 256, 0, stream>>>(Wo, WoT, D, D);

  // 3) QKV projections (Q pre-scaled by 1/sqrt(HD)).
  dim3 ggrid(D / 128, M / 128);  // (16, 64)
  wmma_gemm_kernel<_Float16><<<ggrid, 256, 0, stream>>>(Xq,  WqT, Qh, M, D, D, qscale);
  wmma_gemm_kernel<_Float16><<<ggrid, 256, 0, stream>>>(Xkv, WkT, Kh, M, D, D, 1.0f);
  wmma_gemm_kernel<_Float16><<<ggrid, 256, 0, stream>>>(Xkv, WvT, Vh, M, D, D, 1.0f);

  // 4) Flash attention: grid (q-blocks, heads, batch).
  attn_kernel<<<dim3(S / 128, H, B), 256, 0, stream>>>(Qh, Kh, Vh, Ctx);

  // 5) Output projection to f32.
  wmma_gemm_kernel<float><<<ggrid, 256, 0, stream>>>(Ctx, WoT, out, M, D, D, 1.0f);
}
